// Sdpa_scale_42966852829548
// MI455X (gfx1250) — matrix operand order repaired and checked by reading, not compiled, not run on hardware
//
#include <hip/hip_runtime.h>

typedef __attribute__((ext_vector_type(16))) _Float16 v16h;
typedef __attribute__((ext_vector_type(8)))  _Float16 v8h;
typedef __attribute__((ext_vector_type(8)))  float    v8f;

#define BATCH 32
#define SEQ   2048
#define DHEAD 128
#define BM    128          // query rows per workgroup (8 waves x 16)
#define BN    64           // keys per main-loop step (4 x 16 N-tiles)
#define NWAVE 8
#define KPAD  (DHEAD + 8)  // LDS row padding (halves)
#define VPAD  (BN + 8)

// sched_group_barrier masks
#define SG_WMMA    0x008
#define SG_DSREAD  0x100

// skewed DS-read / WMMA pipeline: 4 reads, 14x(1 wmma + 2 reads), 2 wmma
__device__ __forceinline__ void sched_pipe_16wmma() {
    __builtin_amdgcn_sched_group_barrier(SG_DSREAD, 4, 0);
    #pragma unroll
    for (int i = 0; i < 14; ++i) {
        __builtin_amdgcn_sched_group_barrier(SG_WMMA,   1, 0);
        __builtin_amdgcn_sched_group_barrier(SG_DSREAD, 2, 0);
    }
    __builtin_amdgcn_sched_group_barrier(SG_WMMA, 2, 0);
}

// packed f32x2 -> f16x2 (one v_cvt_pk_rtz_f16_f32)
__device__ __forceinline__ unsigned pk2(float a, float b) {
    auto p = __builtin_amdgcn_cvt_pkrtz(a, b);
    return __builtin_bit_cast(unsigned, p);
}

// ---- 16-lane (half-wave row group) reductions via DPP16: VALU only ----
__device__ __forceinline__ float rowmax16(float x) {
    int t;
    t = __builtin_amdgcn_update_dpp(0, __float_as_int(x), 0xB1,  0xf, 0xf, true); // quad xor1
    x = fmaxf(x, __int_as_float(t));
    t = __builtin_amdgcn_update_dpp(0, __float_as_int(x), 0x4E,  0xf, 0xf, true); // quad xor2
    x = fmaxf(x, __int_as_float(t));
    t = __builtin_amdgcn_update_dpp(0, __float_as_int(x), 0x141, 0xf, 0xf, true); // row_half_mirror
    x = fmaxf(x, __int_as_float(t));
    t = __builtin_amdgcn_update_dpp(0, __float_as_int(x), 0x140, 0xf, 0xf, true); // row_mirror
    x = fmaxf(x, __int_as_float(t));
    return x;
}
__device__ __forceinline__ float rowsum16(float x) {
    int t;
    t = __builtin_amdgcn_update_dpp(0, __float_as_int(x), 0xB1,  0xf, 0xf, true);
    x += __int_as_float(t);
    t = __builtin_amdgcn_update_dpp(0, __float_as_int(x), 0x4E,  0xf, 0xf, true);
    x += __int_as_float(t);
    t = __builtin_amdgcn_update_dpp(0, __float_as_int(x), 0x141, 0xf, 0xf, true);
    x += __int_as_float(t);
    t = __builtin_amdgcn_update_dpp(0, __float_as_int(x), 0x140, 0xf, 0xf, true);
    x += __int_as_float(t);
    return x;
}

__global__ __launch_bounds__(256)
void sdpa_fa_f16wmma(const float* __restrict__ qg,
                     const float* __restrict__ kg,
                     const float* __restrict__ vg,
                     float* __restrict__ og)
{
    __shared__ _Float16 kbuf[2][BN][KPAD];        // [buf][key][d]   row-major
    __shared__ _Float16 vbuf[2][DHEAD][VPAD];     // [buf][d][key]   transposed
    __shared__ _Float16 pbuf[NWAVE][16][VPAD];    // per-wave P tile [m][key]

    const int tid  = threadIdx.x;
    const int wave = tid >> 5;
    const int lane = tid & 31;
    const int ln16 = lane & 15;
    const int hi   = lane >> 4;

    const int qblk = blockIdx.x & 15;             // S / BM = 16
    const int b    = blockIdx.x >> 4;
    const int q0   = qblk * BM;
    const int wrow0 = q0 + wave * 16;             // first query row of this wave

    const size_t base = (size_t)b * SEQ * DHEAD;
    // reference scales by batch size (1/sqrt(32)); fold log2(e) for exp2-domain softmax
    const float qscale = 0.17677669529663687f * 1.4426950408889634f;

    // ---- per-thread staging bases (single address reg + immediate offsets) ----
    const int krow  = tid >> 5;                   // 0..7
    const int kcol  = (tid & 31) * 4;
    const int koff0 = krow * DHEAD + kcol;        // K element offset, i-step = 8*DHEAD
    const int vd    = tid & 127;                  // d (i-invariant)
    const int vk0   = (tid >> 7) * 4;             // 0 or 4
    const int voff0 = vd + vk0 * DHEAD;           // V element offset, (i,j)-step = (8i+j)*DHEAD

    // ---------------- Q tile -> WMMA A-operand registers (scale folded) -----
    v16h a_q[4];
    {
        const float* qrow = qg + base + (size_t)(wrow0 + ln16) * DHEAD + hi * 8;
        #pragma unroll
        for (int c = 0; c < 4; ++c) {
            float tf[16];
            *(float4*)&tf[0]  = *(const float4*)(qrow + 32 * c);
            *(float4*)&tf[4]  = *(const float4*)(qrow + 32 * c + 4);
            *(float4*)&tf[8]  = *(const float4*)(qrow + 32 * c + 16);
            *(float4*)&tf[12] = *(const float4*)(qrow + 32 * c + 20);
            union { v16h h; unsigned u[8]; } qa;
            #pragma unroll
            for (int i = 0; i < 8; ++i)
                qa.u[i] = pk2(tf[2 * i] * qscale, tf[2 * i + 1] * qscale);
            a_q[c] = qa.h;
        }
    }

    // ---------------- softmax state + output accumulators -------------------
    float m_i[8], l_i[8];
    v8f   acc[8];
    #pragma unroll
    for (int r = 0; r < 8; ++r) { m_i[r] = -1e30f; l_i[r] = 0.0f; }
    #pragma unroll
    for (int dt = 0; dt < 8; ++dt)
        #pragma unroll
        for (int r = 0; r < 8; ++r) acc[dt][r] = 0.0f;

    const int ntiles = (qblk + 1) * 2;            // keys 0 .. q0+127, 64 per step

    // prologue: fetch + stage tile 0 into buffer 0
    {
        const float* kt = kg + base + koff0;
        const float* vt = vg + base + voff0;
        _Float16* kdst = &kbuf[0][krow][kcol];
        _Float16* vdst = &vbuf[0][vd][vk0];
        #pragma unroll
        for (int i = 0; i < 8; ++i) {
            const float4 k4 = *(const float4*)(kt + i * (8 * DHEAD));
            uint2 u; u.x = pk2(k4.x, k4.y); u.y = pk2(k4.z, k4.w);
            *(uint2*)(kdst + i * (8 * KPAD)) = u;
        }
        #pragma unroll
        for (int i = 0; i < 8; ++i) {
            float vv[4];
            #pragma unroll
            for (int j = 0; j < 4; ++j) vv[j] = vt[(8 * i + j) * DHEAD];
            uint2 u; u.x = pk2(vv[0], vv[1]); u.y = pk2(vv[2], vv[3]);
            *(uint2*)(vdst + i * 8) = u;
        }
    }

    for (int jt = 0; jt < ntiles; ++jt) {
        const int  cur      = jt & 1;
        const int  kbase    = jt * BN;
        const bool haveNext = (jt + 1 < ntiles);
        const bool gate     = (kbase <= wrow0 + 15);    // per-wave causal gating
        __syncthreads();                  // buf[cur] ready; buf[cur^1] free

        // ==== phase 1 staging: issue next K-tile loads (hidden by QK WMMAs) ====
        float4 kreg[8];
        if (haveNext) {
            const float* kt = kg + base + (size_t)(kbase + BN) * DHEAD + koff0;
            #pragma unroll
            for (int i = 0; i < 8; ++i)
                kreg[i] = *(const float4*)(kt + i * (8 * DHEAD));
            if (jt + 2 < ntiles) {        // GL2 prefetch two tiles ahead
                __builtin_prefetch(kg + base + (size_t)(kbase + 2 * BN) * DHEAD + tid * 32, 0, 2);
                __builtin_prefetch(vg + base + (size_t)(kbase + 2 * BN) * DHEAD + tid * 32, 0, 2);
            }
        }

        // ==== phase 1 compute: S = Q K^T, softmax, P relayout ====
        float alpha[8];
        v16h a_p0, a_p1;                  // P as A operands: keys 0-31, 32-63
        if (gate) {
            const _Float16* kb0 = &kbuf[cur][ln16][hi * 8];

            v8f s[4];
            #pragma unroll
            for (int t = 0; t < 4; ++t) {
                v8f c;
                #pragma unroll
                for (int r = 0; r < 8; ++r) c[r] = 0.0f;
                #pragma unroll
                for (int ch = 0; ch < 4; ++ch) {
                    const _Float16* kp = kb0 + t * (16 * KPAD) + 32 * ch;
                    const v8h blo = *(const v8h*)kp;
                    const v8h bhi = *(const v8h*)(kp + 16);
                    v16h bb;
                    #pragma unroll
                    for (int i = 0; i < 8; ++i) { bb[i] = blo[i]; bb[i + 8] = bhi[i]; }
                    c = __builtin_amdgcn_wmma_f32_16x16x32_f16(
                            false, a_q[ch], false, bb, (short)0, c, false, false);
                }
                s[t] = c;
            }
            sched_pipe_16wmma();          // force DS-read/WMMA software pipeline

            // causal mask only on diagonal-overlap tiles (wave-uniform test)
            if (kbase + BN - 1 > wrow0) {
                #pragma unroll
                for (int t = 0; t < 4; ++t)
                    #pragma unroll
                    for (int r = 0; r < 8; ++r) {
                        const int row = wrow0 + r + hi * 8;
                        const int key = kbase + t * 16 + ln16;
                        s[t][r] = (key > row) ? -1e30f : s[t][r];
                    }
            }

            // online softmax in exp2 domain; DPP row reductions (no LDS traffic)
            #pragma unroll
            for (int r = 0; r < 8; ++r) {
                float mx = fmaxf(fmaxf(s[0][r], s[1][r]), fmaxf(s[2][r], s[3][r]));
                mx = rowmax16(mx);
                const float mnew = fmaxf(m_i[r], mx);
                alpha[r] = exp2f(m_i[r] - mnew);
                #pragma unroll
                for (int t = 0; t < 4; ++t) s[t][r] = exp2f(s[t][r] - mnew);
                const float sum = rowsum16((s[0][r] + s[1][r]) + (s[2][r] + s[3][r]));
                l_i[r] = l_i[r] * alpha[r] + sum;
                m_i[r] = mnew;
            }

            // P (C layout) -> per-wave LDS -> A layout; wave-internal ordering only
            {
                _Float16* pw = &pbuf[wave][hi * 8][ln16];     // + r*VPAD + t*16 imm
                #pragma unroll
                for (int r = 0; r < 8; ++r)
                    #pragma unroll
                    for (int t = 0; t < 4; ++t)
                        pw[r * VPAD + t * 16] = (_Float16)s[t][r];
            }
            __threadfence_block();
            __builtin_amdgcn_wave_barrier();

            {
                const _Float16* pp = &pbuf[wave][ln16][hi * 8];
                const v8h p0 = *(const v8h*)pp;
                const v8h p1 = *(const v8h*)(pp + 16);
                const v8h p2 = *(const v8h*)(pp + 32);
                const v8h p3 = *(const v8h*)(pp + 48);
                #pragma unroll
                for (int i = 0; i < 8; ++i) {
                    a_p0[i] = p0[i]; a_p0[i + 8] = p1[i];
                    a_p1[i] = p2[i]; a_p1[i + 8] = p3[i];
                }
            }
        }

        // ==== retire K staging (kreg dies here) ====
        if (haveNext) {
            _Float16* kdst = &kbuf[cur ^ 1][krow][kcol];
            #pragma unroll
            for (int i = 0; i < 8; ++i) {
                uint2 u; u.x = pk2(kreg[i].x, kreg[i].y); u.y = pk2(kreg[i].z, kreg[i].w);
                *(uint2*)(kdst + i * (8 * KPAD)) = u;
            }
        }

        // ==== phase 2 staging: issue next V-tile loads (hidden by PV WMMAs) ====
        float vreg[32];
        if (haveNext) {
            const float* vt = vg + base + (size_t)(kbase + BN) * DHEAD + voff0;
            #pragma unroll
            for (int i = 0; i < 8; ++i)
                #pragma unroll
                for (int j = 0; j < 4; ++j)
                    vreg[i * 4 + j] = vt[(8 * i + j) * DHEAD];
        }

        // ==== phase 2 compute: O = diag(alpha)·O + P·V over 8 D-tiles ====
        if (gate) {
            const _Float16* vb0 = &vbuf[cur][ln16][hi * 8];

            // rescale accumulators (fills scheduler gaps in the WMMA pipeline)
            #pragma unroll
            for (int dt = 0; dt < 8; ++dt)
                #pragma unroll
                for (int r = 0; r < 8; ++r) acc[dt][r] *= alpha[r];

            #pragma unroll
            for (int dt = 0; dt < 8; ++dt) {
                const _Float16* vp = vb0 + dt * (16 * VPAD);
                v16h b0, b1;
                {
                    const v8h l0 = *(const v8h*)vp,        h0 = *(const v8h*)(vp + 16);
                    const v8h l1 = *(const v8h*)(vp + 32), h1 = *(const v8h*)(vp + 48);
                    #pragma unroll
                    for (int i = 0; i < 8; ++i) {
                        b0[i] = l0[i]; b0[i + 8] = h0[i];
                        b1[i] = l1[i]; b1[i + 8] = h1[i];
                    }
                }
                acc[dt] = __builtin_amdgcn_wmma_f32_16x16x32_f16(
                              false, a_p0, false, b0, (short)0, acc[dt], false, false);
                acc[dt] = __builtin_amdgcn_wmma_f32_16x16x32_f16(
                              false, a_p1, false, b1, (short)0, acc[dt], false, false);
            }
            sched_pipe_16wmma();          // force DS-read/WMMA software pipeline
        }

        // ==== retire V staging (vreg dies here) ====
        if (haveNext) {
            _Float16* vdst = &vbuf[cur ^ 1][vd][vk0];
            #pragma unroll
            for (int i = 0; i < 8; ++i) {
                uint2 u; u.x = pk2(vreg[i * 4], vreg[i * 4 + 1]);
                         u.y = pk2(vreg[i * 4 + 2], vreg[i * 4 + 3]);
                *(uint2*)(vdst + i * 8) = u;
            }
        }
    }

    // ---------------- epilogue: O / l ----------------
    {
        float* orow = og + base + (size_t)(wrow0 + hi * 8) * DHEAD + ln16;
        #pragma unroll
        for (int dt = 0; dt < 8; ++dt)
            #pragma unroll
            for (int r = 0; r < 8; ++r)
                orow[r * DHEAD + dt * 16] = acc[dt][r] / l_i[r];
    }
}

extern "C" void kernel_launch(void* const* d_in, const int* in_sizes, int n_in,
                              void* d_out, int out_size, void* d_ws, size_t ws_size,
                              hipStream_t stream) {
    const float* q = (const float*)d_in[0];
    const float* k = (const float*)d_in[1];
    const float* v = (const float*)d_in[2];
    float* out = (float*)d_out;
    (void)in_sizes; (void)n_in; (void)out_size; (void)d_ws; (void)ws_size;

    dim3 grid(BATCH * (SEQ / BM));   // 32 * 16 = 512 workgroups
    dim3 block(256);                 // 8 waves (wave32)
    sdpa_fa_f16wmma<<<grid, block, 0, stream>>>(q, k, v, out);
}
